// RGBuvHistBlock_1889785610949
// MI455X (gfx1250) — matrix operand order repaired and checked by reading, not compiled, not run on hardware
//
#include <hip/hip_runtime.h>
#include <hip/hip_bf16.h>

typedef __attribute__((ext_vector_type(16))) _Float16 v16h;
typedef __attribute__((ext_vector_type(8)))  float    v8f;

#define EPS_F      1e-06f
#define H_BINS     64
#define SIGMA_F    0.02f
#define NPIX       22500          // 150*150
#define NCHUNK     704            // ceil(22500/32); 704*32 = 22528
#define SLICES     8
#define OUT_ELEMS  393216         // 32*3*64*64
#define PER_BATCH  12288          // 3*64*64

// ---------------------------------------------------------------------------
// Kernel 0: zero the accumulation buffers (d_out is atomically accumulated,
// d_ws[b] holds per-batch normalization sums; both are poisoned by harness).
// ---------------------------------------------------------------------------
__global__ __launch_bounds__(256) void zero_kernel(float* __restrict__ out,
                                                   float* __restrict__ ws) {
    int i = blockIdx.x * 256 + threadIdx.x;
    if (i < OUT_ELEMS) out[i] = 0.0f;
    if (i < 32)        ws[i]  = 0.0f;
}

// ---------------------------------------------------------------------------
// Kernel 1: per-(b,c) unnormalized 64x64 UV histogram via WMMA.
//   hist[u,v] = sum_n Iy[n]*Ku[n,u]*Kv[n,v]  ==  (Iy*Ku)^T @ Kv
// M=N=64 (4x4 tiles of 16x16), K = pixels in chunks of 32 (f16 WMMA K).
// ---------------------------------------------------------------------------
__global__ __launch_bounds__(256) void hist_kernel(const float* __restrict__ x,
                                                   float* __restrict__ out,
                                                   float* __restrict__ bsum) {
    const int gid   = blockIdx.x;          // 0 .. 96*SLICES-1
    const int slice = gid % SLICES;
    const int bc    = gid / SLICES;        // 0..95
    const int b     = bc / 3;
    const int c     = bc % 3;

    const int lane = threadIdx.x & 31;
    const int wave = threadIdx.x >> 5;     // 0..7
    const int l15  = lane & 15;
    const int hi16 = lane >> 4;            // 0 or 1

    const float inv_sig2 = 1.0f / (SIGMA_F * SIGMA_F);   // 2500

    // bin value owned by this lane, for each of the 4 tile rows/cols
    float binm[4];
#pragma unroll
    for (int t = 0; t < 4; ++t)
        binm[t] = -3.0f + 6.0f * (float)(16 * t + l15) * (1.0f / 63.0f);

    const float* __restrict__ xb = x + (size_t)b * 3u * NPIX;

    v8f acc[4][4];
#pragma unroll
    for (int mi = 0; mi < 4; ++mi)
#pragma unroll
        for (int ni = 0; ni < 4; ++ni)
            acc[mi][ni] = (v8f){0.f, 0.f, 0.f, 0.f, 0.f, 0.f, 0.f, 0.f};

    // NCHUNK == 704 == 64*11 : every (slice, wave) pair gets exactly 11 chunks,
    // so the chunk loop is wave-uniform (EXEC all-ones at every WMMA).
    for (int ch = slice * 8 + wave; ch < NCHUNK; ch += SLICES * 8) {
        // ---- this lane owns pixel (ch*32 + lane) of image b ----
        int   p  = ch * 32 + lane;
        bool  ok = (p < NPIX);
        int   pc = ok ? p : 0;
        float r  = xb[0 * NPIX + pc];
        float g  = xb[1 * NPIX + pc];
        float bl = xb[2 * NPIX + pc];
        r  = fminf(fmaxf(r, 0.0f), 1.0f);
        g  = fminf(fmaxf(g, 0.0f), 1.0f);
        bl = fminf(fmaxf(bl, 0.0f), 1.0f);
        float iy = ok ? __builtin_sqrtf(r * r + g * g + bl * bl + EPS_F) : 0.0f;
        float lr = __logf(r + EPS_F);
        float lg = __logf(g + EPS_F);
        float lb = __logf(bl + EPS_F);
        float uval, vval;
        if (c == 0)      { uval = lr - lg; vval = lr - lb; }
        else if (c == 1) { uval = lg - lr; vval = lg - lb; }
        else             { uval = lb - lr; vval = lb - lg; }

        // ---- gather the K-slots this lane needs (ISA fragment layouts) ----
        float uA[16], iyA[16], vB[16];
#pragma unroll
        for (int h = 0; h < 16; ++h) {
            int kA = (h & 7) + ((h >> 3) << 4) + (hi16 << 3);  // A 16x32 f16
            int kB = (h & 7) + ((h >> 3) << 4) + (hi16 << 3);  // B 32x16 f16
            uA[h]  = __shfl(uval, kA, 32);
            iyA[h] = __shfl(iy,   kA, 32);
            vB[h]  = __shfl(vval, kB, 32);
        }

        // ---- B fragments: Kv(k, n-bin), n = 16*ni + (lane&15) ----
        v16h bfrag[4];
#pragma unroll
        for (int ni = 0; ni < 4; ++ni) {
#pragma unroll
            for (int h = 0; h < 16; ++h) {
                float d  = vB[h] - binm[ni];
                float kv = __builtin_amdgcn_rcpf(__builtin_fmaf(d * d, inv_sig2, 1.0f));
                bfrag[ni][h] = (_Float16)kv;
            }
        }

        // ---- A fragments (one at a time): Iy(k)*Ku(k, m-bin), then WMMA ----
#pragma unroll
        for (int mi = 0; mi < 4; ++mi) {
            v16h afrag;
#pragma unroll
            for (int h = 0; h < 16; ++h) {
                float d  = uA[h] - binm[mi];
                float ku = __builtin_amdgcn_rcpf(__builtin_fmaf(d * d, inv_sig2, 1.0f));
                afrag[h] = (_Float16)(iyA[h] * ku);
            }
#pragma unroll
            for (int ni = 0; ni < 4; ++ni) {
                acc[mi][ni] = __builtin_amdgcn_wmma_f32_16x16x32_f16(
                    /*neg_a=*/false, afrag,
                    /*neg_b=*/false, bfrag[ni],
                    /*c_mod=*/(short)0, acc[mi][ni],
                    /*reuse_a=*/false, /*reuse_b=*/false);
            }
        }
    }

    // ---- reduce the 8 waves' accumulators in LDS ----
    __shared__ float lhist[H_BINS * H_BINS];
    for (int i = threadIdx.x; i < H_BINS * H_BINS; i += 256) lhist[i] = 0.0f;
    __syncthreads();

#pragma unroll
    for (int mi = 0; mi < 4; ++mi)
#pragma unroll
        for (int ni = 0; ni < 4; ++ni)
#pragma unroll
            for (int rr = 0; rr < 8; ++rr) {
                int u = 16 * mi + rr + 8 * hi16;   // D-matrix f32 layout
                int v = 16 * ni + l15;
                atomicAdd(&lhist[u * H_BINS + v], acc[mi][ni][rr]);
            }
    __syncthreads();

    // ---- accumulate into global hist + per-batch normalization sum ----
    float psum = 0.0f;
    float* __restrict__ outbc = out + (size_t)bc * (H_BINS * H_BINS);
    for (int i = threadIdx.x; i < H_BINS * H_BINS; i += 256) {
        float hv = lhist[i];
        atomicAdd(&outbc[i], hv);
        psum += hv;
    }
#pragma unroll
    for (int off = 16; off > 0; off >>= 1)
        psum += __shfl_down(psum, off, 32);

    __shared__ float wsum[8];
    if (lane == 0) wsum[wave] = psum;
    __syncthreads();
    if (threadIdx.x == 0) {
        float s = 0.0f;
#pragma unroll
        for (int w = 0; w < 8; ++w) s += wsum[w];
        atomicAdd(&bsum[b], s);
    }
}

// ---------------------------------------------------------------------------
// Kernel 2: normalize by per-batch total (+ EPS)
// ---------------------------------------------------------------------------
__global__ __launch_bounds__(256) void norm_kernel(float* __restrict__ out,
                                                   const float* __restrict__ bsum) {
    int i = blockIdx.x * 256 + threadIdx.x;
    if (i < OUT_ELEMS) {
        int b = i / PER_BATCH;
        out[i] = out[i] * __builtin_amdgcn_rcpf(bsum[b] + EPS_F);
    }
}

extern "C" void kernel_launch(void* const* d_in, const int* in_sizes, int n_in,
                              void* d_out, int out_size, void* d_ws, size_t ws_size,
                              hipStream_t stream) {
    (void)in_sizes; (void)n_in; (void)out_size; (void)ws_size;
    const float* x   = (const float*)d_in[0];
    float*       out = (float*)d_out;
    float*       ws  = (float*)d_ws;

    zero_kernel<<<(OUT_ELEMS + 255) / 256, 256, 0, stream>>>(out, ws);
    hist_kernel<<<96 * SLICES, 256, 0, stream>>>(x, out, ws);
    norm_kernel<<<(OUT_ELEMS + 255) / 256, 256, 0, stream>>>(out, ws);
}
